// GCN_1520418423141
// MI455X (gfx1250) — matrix operand order repaired and checked by reading, not compiled, not run on hardware
//
#include <hip/hip_runtime.h>

#define NNODES 50000
#define NDIN   256
#define NDOUT  256
#define NEDGE  800000

typedef __bf16 bf16_t;
typedef __attribute__((ext_vector_type(16))) __bf16 v16bf;
typedef __attribute__((ext_vector_type(8)))  float  v8f;
typedef __attribute__((ext_vector_type(4)))  unsigned int v4u;

union Frag16 { v16bf v; v4u q[2]; };

__device__ __forceinline__ unsigned int sage_hash(unsigned int x) {
    x ^= x >> 16; x *= 0x7feb352dU;
    x ^= x >> 15; x *= 0x846ca68bU;
    x ^= x >> 16;
    return x;
}

// ---- zero agg + cnt ---------------------------------------------------------
__global__ void __launch_bounds__(256) sage_zero(float* __restrict__ agg,
                                                 float* __restrict__ cnt) {
    int i = blockIdx.x * 256 + threadIdx.x;
    if (i < NNODES * NDIN) agg[i] = 0.0f;
    if (i < NNODES)        cnt[i] = 0.0f;
}

// ---- dropout: f32 -> bf16, deterministic hash mask, scale 1/(1-p)=2 ---------
__global__ void __launch_bounds__(256) sage_dropout(const float* __restrict__ x,
                                                    bf16_t* __restrict__ xd) {
    int i = blockIdx.x * 256 + threadIdx.x;
    if (i >= NNODES * NDIN) return;
    unsigned int h = sage_hash((unsigned int)i ^ 0x9e3779b9u);
    float v = (h >> 31) ? x[i] * 2.0f : 0.0f;
    xd[i] = (bf16_t)v;
}

// ---- in-degree counts -------------------------------------------------------
__global__ void __launch_bounds__(256) sage_degree(const int* __restrict__ dst,
                                                   float* __restrict__ cnt) {
    int e = blockIdx.x * 256 + threadIdx.x;
    if (e < NEDGE) atomicAdd(&cnt[dst[e]], 1.0f);
}

// ---- edge scatter: agg[dst] += xd[src] (feature-parallel, coalesced) --------
// bf16 gather halves read traffic; f32 atomic adds hit the 192MB L2 (agg is
// 51.2MB resident). This phase is the roofline bottleneck (~1.2GB L2 traffic).
__global__ void __launch_bounds__(256) sage_scatter(const int* __restrict__ src,
                                                    const int* __restrict__ dst,
                                                    const bf16_t* __restrict__ xd,
                                                    float* __restrict__ agg) {
    int f  = threadIdx.x;            // feature lane 0..255
    int e0 = blockIdx.x * 4;         // 4 edges per block
#pragma unroll
    for (int i = 0; i < 4; ++i) {
        int e = e0 + i;
        if (e >= NEDGE) break;
        int s = src[e];
        int d = dst[e];
        if (i < 3) __builtin_prefetch(&xd[(size_t)src[e + 1] * NDIN + f], 0, 1);
        float v = (float)xd[(size_t)s * NDIN + f];
        atomicAdd(&agg[(size_t)d * NDIN + f], v);
    }
}

// ---- mean = agg / max(cnt,1), stored bf16 for WMMA A-fragments --------------
__global__ void __launch_bounds__(256) sage_mean(const float* __restrict__ agg,
                                                 const float* __restrict__ cnt,
                                                 bf16_t* __restrict__ mn) {
    int i = blockIdx.x * 256 + threadIdx.x;
    if (i >= NNODES * NDIN) return;
    float c = cnt[i / NDIN];
    mn[i] = (bf16_t)(agg[i] / fmaxf(c, 1.0f));
}

// ---- pack W_l / W_r into WMMA B-fragment order ------------------------------
// Bp[w][ktile(8)][ntile(16)][lane(32)][v(8)] dwords; dword packs
// bf16(W[k][n]) | bf16(W[k+1][n])<<16, k = ktile*32 + 8*(lane>>4) + 2*v + 8*(v>=4),
// n = ntile*16 + (lane&15) — per ISA 16-bit B-matrix VGPR layout.
__global__ void __launch_bounds__(256) sage_packB(const float* __restrict__ Wl,
                                                  const float* __restrict__ Wr,
                                                  unsigned int* __restrict__ Bp) {
    int id = blockIdx.x * 256 + threadIdx.x;
    if (id >= 2 * 8 * 16 * 32) return;
    int lane = id & 31;
    int t = id >> 5;
    int ntile = t & 15; t >>= 4;
    int ktile = t & 7;
    int w = t >> 3;
    const float* W = w ? Wr : Wl;
    unsigned int* o = Bp + (size_t)w * 32768 + (((ktile * 16 + ntile) * 32) + lane) * 8;
    int half = lane >> 4;
    int n = ntile * 16 + (lane & 15);
#pragma unroll
    for (int v = 0; v < 8; ++v) {
        int k = ktile * 32 + half * 8 + 2 * v + (v >= 4 ? 8 : 0);
        union { bf16_t b; unsigned short u; } lo, hi;
        lo.b = (bf16_t)W[(size_t)k * NDOUT + n];
        hi.b = (bf16_t)W[(size_t)(k + 1) * NDOUT + n];
        o[v] = (unsigned int)lo.u | ((unsigned int)hi.u << 16);
    }
}

__device__ __forceinline__ void ld_frag_a(Frag16& a, const bf16_t* pa) {
    a.q[0] = *(const v4u*)(pa);
    a.q[1] = *(const v4u*)(pa + 16);
}
__device__ __forceinline__ void ld_frag_b(Frag16& b, const unsigned int* pb) {
    b.q[0] = *(const v4u*)(pb);
    b.q[1] = *(const v4u*)(pb + 4);
}

// ---- fused WMMA GEMM: out = mean@W_l + xd@W_r + b_l -------------------------
// Wave -> 16x64 output tile (4 f32 accumulators). Both K=256 passes flattened
// into one 16-step K pipeline with double-buffered fragments so the scheduler
// can hoist step s+1 loads above step s WMMAs (relaxed s_wait_loadcnt instead
// of hard ==0 stalls). Output via non-temporal stores (write-once stream).
__global__ void __launch_bounds__(256) sage_gemm(const bf16_t* __restrict__ Amean,
                                                 const bf16_t* __restrict__ Axd,
                                                 const unsigned int* __restrict__ Bp,
                                                 const float* __restrict__ bias,
                                                 float* __restrict__ out) {
    int wave  = threadIdx.x >> 5;
    int lane  = threadIdx.x & 31;
    int strip = blockIdx.x * 8 + wave;       // 16-row strip
    if (strip * 16 >= NNODES) return;        // wave-uniform: EXEC stays all-1s
    int m0   = strip * 16;
    int n0   = blockIdx.y * 64;
    int half = lane >> 4;
    int m    = m0 + (lane & 15);
    int nt0  = n0 >> 4;

    const bf16_t* Abase0 = Amean + (size_t)m * NDIN + 8 * half;
    const bf16_t* Abase1 = Axd   + (size_t)m * NDIN + 8 * half;
    const unsigned int* Blane = Bp + ((nt0 * 32) + lane) * 8;

    v8f acc0 = {}, acc1 = {}, acc2 = {}, acc3 = {};

    Frag16 a[2];
    Frag16 b[2][4];

    // step s in [0,16): pass = s>>3 selects (mean,W_l) vs (xd,W_r); kt = s&7
    ld_frag_a(a[0], Abase0);
    ld_frag_b(b[0][0], Blane + 0);
    ld_frag_b(b[0][1], Blane + 256);
    ld_frag_b(b[0][2], Blane + 512);
    ld_frag_b(b[0][3], Blane + 768);

#pragma unroll
    for (int s = 0; s < 16; ++s) {
        int cur = s & 1;
        int nxt = cur ^ 1;
        if (s < 15) {
            int sn   = s + 1;
            int kt   = sn & 7;
            const bf16_t* pa = ((sn >> 3) ? Abase1 : Abase0) + kt * 32;
            const unsigned int* pb = Blane + (size_t)(sn >> 3) * 32768 + kt * (16 * 32 * 8);
            ld_frag_a(a[nxt], pa);
            ld_frag_b(b[nxt][0], pb + 0);
            ld_frag_b(b[nxt][1], pb + 256);
            ld_frag_b(b[nxt][2], pb + 512);
            ld_frag_b(b[nxt][3], pb + 768);
        }
        acc0 = __builtin_amdgcn_wmma_f32_16x16x32_bf16(false, a[cur].v, false, b[cur][0].v,
                                                       (short)0, acc0, false, false);
        acc1 = __builtin_amdgcn_wmma_f32_16x16x32_bf16(false, a[cur].v, false, b[cur][1].v,
                                                       (short)0, acc1, false, false);
        acc2 = __builtin_amdgcn_wmma_f32_16x16x32_bf16(false, a[cur].v, false, b[cur][2].v,
                                                       (short)0, acc2, false, false);
        acc3 = __builtin_amdgcn_wmma_f32_16x16x32_bf16(false, a[cur].v, false, b[cur][3].v,
                                                       (short)0, acc3, false, false);
    }

    // C/D layout: VGPR r -> row m0 + r + 8*half, col n = n_tile + (lane&15)
    int nb = n0 + (lane & 15);
    {
        float bv = bias[nb];
#pragma unroll
        for (int r = 0; r < 8; ++r)
            __builtin_nontemporal_store(acc0[r] + bv,
                &out[(size_t)(m0 + r + 8 * half) * NDOUT + nb]);
    }
    {
        float bv = bias[nb + 16];
#pragma unroll
        for (int r = 0; r < 8; ++r)
            __builtin_nontemporal_store(acc1[r] + bv,
                &out[(size_t)(m0 + r + 8 * half) * NDOUT + nb + 16]);
    }
    {
        float bv = bias[nb + 32];
#pragma unroll
        for (int r = 0; r < 8; ++r)
            __builtin_nontemporal_store(acc2[r] + bv,
                &out[(size_t)(m0 + r + 8 * half) * NDOUT + nb + 32]);
    }
    {
        float bv = bias[nb + 48];
#pragma unroll
        for (int r = 0; r < 8; ++r)
            __builtin_nontemporal_store(acc3[r] + bv,
                &out[(size_t)(m0 + r + 8 * half) * NDOUT + nb + 48]);
    }
}

extern "C" void kernel_launch(void* const* d_in, const int* in_sizes, int n_in,
                              void* d_out, int out_size, void* d_ws, size_t ws_size,
                              hipStream_t stream) {
    (void)in_sizes; (void)n_in; (void)out_size; (void)ws_size;

    const float* x  = (const float*)d_in[0];
    const int*   ei = (const int*)d_in[1];
    const float* Wl = (const float*)d_in[2];
    const float* bl = (const float*)d_in[3];
    const float* Wr = (const float*)d_in[4];
    float* out = (float*)d_out;

    const int* src = ei;
    const int* dst = ei + NEDGE;

    char* ws = (char*)d_ws;
    size_t o = 0;
    bf16_t* xd  = (bf16_t*)(ws + o); o += (size_t)NNODES * NDIN * 2;   // 25.6 MB
    bf16_t* mn  = (bf16_t*)(ws + o); o += (size_t)NNODES * NDIN * 2;   // 25.6 MB
    float*  agg = (float*) (ws + o); o += (size_t)NNODES * NDIN * 4;   // 51.2 MB
    float*  cnt = (float*) (ws + o); o += ((size_t)NNODES * 4 + 255) & ~(size_t)255;
    unsigned int* Bp = (unsigned int*)(ws + o);                         // 256 KB

    const int elems = NNODES * NDIN;

    sage_zero   <<<(elems + 255) / 256, 256, 0, stream>>>(agg, cnt);
    sage_dropout<<<(elems + 255) / 256, 256, 0, stream>>>(x, xd);
    sage_degree <<<(NEDGE + 255) / 256, 256, 0, stream>>>(dst, cnt);
    sage_scatter<<<(NEDGE + 3) / 4,     256, 0, stream>>>(src, dst, xd, agg);
    sage_mean   <<<(elems + 255) / 256, 256, 0, stream>>>(agg, cnt, mn);
    sage_packB  <<<32,                  256, 0, stream>>>(Wl, Wr, Bp);

    dim3 ggrid((NNODES / 16 + 7) / 8, NDOUT / 64);  // (391, 4)
    sage_gemm<<<ggrid, 256, 0, stream>>>(mn, xd, Bp, bl, out);
}
